// MoEMLP_30202210025562
// MI455X (gfx1250) — matrix operand order repaired and checked by reading, not compiled, not run on hardware
//
#include <hip/hip_runtime.h>

#define DD 1024
#define EE 64
#define HH 16
#define NTOK 32768
#define MB 64           // tokens per workgroup (4 M-tiles of 16)
#define BLK 256         // 8 waves

typedef __attribute__((ext_vector_type(16))) __bf16 bf16x16;
typedef __attribute__((ext_vector_type(8)))  float  f32x8;

union FragU { bf16x16 v; int4 q[2]; };

// A-matrix 16x32 bf16 fragment from row-major [16][DD] (ISA 7.12.2 layout):
// lane grp 0: v0-3 = K(base+0..7), v4-7 = K(base+16..23); grp 1: +8
__device__ __forceinline__ bf16x16 load_afrag(const __bf16* base, int kt, int grp, int m) {
  const __bf16* p = base + m * DD + kt * 32 + grp * 8;
  FragU f;
  f.q[0] = *(const int4*)(p);
  f.q[1] = *(const int4*)(p + 16);
  return f.v;
}

// B-matrix 32x16 bf16 fragment from B^T row-major [16][DD]:
// lane holds column n = lane&15; v0-3 = K(grp*8+0..7), v4-7 = K(16+grp*8+0..7)
__device__ __forceinline__ bf16x16 load_bfrag(const __bf16* btile, int kt, int grp, int n) {
  const __bf16* p = btile + (size_t)n * DD + kt * 32 + grp * 8;
  FragU f;
  f.q[0] = *(const int4*)(p);
  f.q[1] = *(const int4*)(p + 16);
  return f.v;
}

#define WMMA_BF16(A, B, C) \
  __builtin_amdgcn_wmma_f32_16x16x32_bf16(false, (A), false, (B), (short)0, (C), false, false)

// K=1024 GEMM: one shared B tile, 4 M-tile accumulators (4x B reuse per fetch)
__device__ __forceinline__ void gemm4(const __bf16* As /* LDS [64][DD] */,
                                      const __bf16* Bt /* B^T tile [16][DD] */,
                                      int grp, int ln,
                                      f32x8& c0, f32x8& c1, f32x8& c2, f32x8& c3) {
  #pragma unroll 2
  for (int kt = 0; kt < 32; ++kt) {
    bf16x16 b  = load_bfrag(Bt, kt, grp, ln);
    bf16x16 a0 = load_afrag(As + 0 * 16 * DD, kt, grp, ln);
    c0 = WMMA_BF16(a0, b, c0);
    bf16x16 a1 = load_afrag(As + 1 * 16 * DD, kt, grp, ln);
    c1 = WMMA_BF16(a1, b, c1);
    bf16x16 a2 = load_afrag(As + 2 * 16 * DD, kt, grp, ln);
    c2 = WMMA_BF16(a2, b, c2);
    bf16x16 a3 = load_afrag(As + 3 * 16 * DD, kt, grp, ln);
    c3 = WMMA_BF16(a3, b, c3);
  }
}

// ---- preprocessing: fp32 -> bf16 transposed weight copies in workspace ----

// w1 [E][D][H] -> w1t [E][H][D]   (B^T per expert: rows = H, K-contiguous)
__global__ __launch_bounds__(256) void moe_prep_w1(const float* __restrict__ w1,
                                                   __bf16* __restrict__ w1t) {
  int i = blockIdx.x * 256 + threadIdx.x;
  if (i >= EE * HH * DD) return;
  int d = i % DD;
  int t = i / DD;
  int h = t % HH;
  int e = t / HH;
  w1t[i] = (__bf16)w1[(size_t)e * DD * HH + (size_t)d * HH + h];
}

// w2 [E][H][D] (== W2flat [K=E*H][D]) -> w2t [D][E*H]  (B^T: rows = D, K-contiguous)
__global__ __launch_bounds__(256) void moe_prep_w2(const float* __restrict__ w2,
                                                   __bf16* __restrict__ w2t) {
  int i = blockIdx.x * 256 + threadIdx.x;
  if (i >= DD * EE * HH) return;
  int k = i % (EE * HH);
  int d = i / (EE * HH);
  w2t[i] = (__bf16)w2[(size_t)k * DD + d];
}

// gate_w [D][E] -> gwt [E][D]  (B^T: rows = E, K-contiguous)
__global__ __launch_bounds__(256) void moe_prep_gw(const float* __restrict__ gw,
                                                   __bf16* __restrict__ gwt) {
  int i = blockIdx.x * 256 + threadIdx.x;
  if (i >= EE * DD) return;
  int d = i % DD;
  int e = i / DD;
  gwt[i] = (__bf16)gw[(size_t)d * EE + e];
}

// ---- main fused MoE kernel: 1 block (8 waves) per 64-token block ----

__global__ __launch_bounds__(BLK) void moe_main(
    const float* __restrict__ x,       // [N][D] fp32
    const float* __restrict__ gate_b,  // [E]
    const float* __restrict__ b1,      // [E][H]
    const __bf16* __restrict__ gwt,    // [E][D]
    const __bf16* __restrict__ w1t,    // [E][H][D]
    const __bf16* __restrict__ w2t,    // [D][E*H]
    float* __restrict__ out)           // [N][D] fp32
{
  __shared__ __bf16 xs[MB * DD];       // 128 KB: x block, bf16
  __shared__ __bf16 hs[MB * DD];       // 128 KB: gate-scaled relu(h), bf16
  __shared__ float  gs[EE * MB];       // 16 KB: gate probs, [expert][token]

  const int row0 = blockIdx.x * MB;
  const int tid  = threadIdx.x;
  const int wave = tid >> 5;
  const int lane = tid & 31;
  const int grp  = lane >> 4;
  const int ln   = lane & 15;

  // Phase 1: cooperative fp32 -> bf16 load of the x block into LDS
  {
    const float4* xin = (const float4*)(x + (size_t)row0 * DD);
    #pragma unroll 4
    for (int i = tid; i < MB * DD / 4; i += BLK) {
      float4 v = xin[i];
      __bf16* p = &xs[i * 4];
      p[0] = (__bf16)v.x; p[1] = (__bf16)v.y;
      p[2] = (__bf16)v.z; p[3] = (__bf16)v.w;
    }
  }
  __syncthreads();

  // Phase 2: gating; waves 0..3 each handle one M-tile (wave == M-tile index)
  if (wave < 4) {
    const __bf16* axs = xs + wave * 16 * DD;
    f32x8 lt0 = {}, lt1 = {}, lt2 = {}, lt3 = {};
    #pragma unroll 2
    for (int kt = 0; kt < 32; ++kt) {
      bf16x16 a  = load_afrag(axs, kt, grp, ln);
      bf16x16 g0 = load_bfrag(gwt +  0 * DD, kt, grp, ln);
      lt0 = WMMA_BF16(a, g0, lt0);
      bf16x16 g1 = load_bfrag(gwt + 16 * DD, kt, grp, ln);
      lt1 = WMMA_BF16(a, g1, lt1);
      bf16x16 g2 = load_bfrag(gwt + 32 * DD, kt, grp, ln);
      lt2 = WMMA_BF16(a, g2, lt2);
      bf16x16 g3 = load_bfrag(gwt + 48 * DD, kt, grp, ln);
      lt3 = WMMA_BF16(a, g3, lt3);
    }
    float gb0 = gate_b[ 0 + ln], gb1 = gate_b[16 + ln];
    float gb2 = gate_b[32 + ln], gb3 = gate_b[48 + ln];
    #pragma unroll
    for (int r = 0; r < 8; ++r) {
      lt0[r] += gb0; lt1[r] += gb1; lt2[r] += gb2; lt3[r] += gb3;
    }
    // per-token softmax over 64 experts (4 tiles x 16 lanes of each half-wave)
    f32x8 mx;
    #pragma unroll
    for (int r = 0; r < 8; ++r)
      mx[r] = fmaxf(fmaxf(lt0[r], lt1[r]), fmaxf(lt2[r], lt3[r]));
    #pragma unroll
    for (int off = 8; off >= 1; off >>= 1)
      #pragma unroll
      for (int r = 0; r < 8; ++r)
        mx[r] = fmaxf(mx[r], __shfl_xor(mx[r], off, 32));
    f32x8 sum = {};
    #pragma unroll
    for (int r = 0; r < 8; ++r) {
      lt0[r] = __expf(lt0[r] - mx[r]);
      lt1[r] = __expf(lt1[r] - mx[r]);
      lt2[r] = __expf(lt2[r] - mx[r]);
      lt3[r] = __expf(lt3[r] - mx[r]);
      sum[r] = lt0[r] + lt1[r] + lt2[r] + lt3[r];
    }
    #pragma unroll
    for (int off = 8; off >= 1; off >>= 1)
      #pragma unroll
      for (int r = 0; r < 8; ++r)
        sum[r] += __shfl_xor(sum[r], off, 32);
    // scatter probs: gs[expert][token], token = wave*16 + grp*8 + r
    const int t0 = wave * 16 + grp * 8;
    #pragma unroll
    for (int r = 0; r < 8; ++r) {
      gs[( 0 + ln) * MB + t0 + r] = lt0[r] / sum[r];
      gs[(16 + ln) * MB + t0 + r] = lt1[r] / sum[r];
      gs[(32 + ln) * MB + t0 + r] = lt2[r] / sum[r];
      gs[(48 + ln) * MB + t0 + r] = lt3[r] / sum[r];
    }
  }
  __syncthreads();

  // Phase 3: wave's 8 experts, all 4 M-tiles: h = relu(x@w1[e]+b1[e]) * g[:,e]
  for (int j = 0; j < 8; ++j) {
    const int e = wave * 8 + j;
    const __bf16* bt = w1t + (size_t)e * HH * DD;  // [16][1024] B^T
    f32x8 c0 = {}, c1 = {}, c2 = {}, c3 = {};
    gemm4(xs, bt, grp, ln, c0, c1, c2, c3);
    const float bias = b1[e * HH + ln];
    const float* ge = gs + e * MB;
    #pragma unroll
    for (int mt = 0; mt < 4; ++mt) {
      f32x8 c = (mt == 0) ? c0 : (mt == 1) ? c1 : (mt == 2) ? c2 : c3;
      const int t0 = mt * 16 + grp * 8;
      #pragma unroll
      for (int r = 0; r < 8; ++r) {
        float sc = ge[t0 + r];  // broadcast LDS read (uniform per 16-lane group)
        float v  = fmaxf(c[r] + bias, 0.0f) * sc;
        hs[(t0 + r) * DD + e * 16 + ln] = (__bf16)v;
      }
    }
  }
  __syncthreads();

  // Phase 4: out[64][1024] = hs @ W2flat; wave owns 128 output columns
  for (int nt = 0; nt < 8; ++nt) {
    const int n0 = wave * 128 + nt * 16;
    const __bf16* bt = w2t + (size_t)n0 * DD;  // rows n0..n0+15 of B^T (stride E*H == DD)
    f32x8 c0 = {}, c1 = {}, c2 = {}, c3 = {};
    gemm4(hs, bt, grp, ln, c0, c1, c2, c3);
    #pragma unroll
    for (int mt = 0; mt < 4; ++mt) {
      f32x8 c = (mt == 0) ? c0 : (mt == 1) ? c1 : (mt == 2) ? c2 : c3;
      float* op = out + (size_t)(row0 + mt * 16 + grp * 8) * DD + n0 + ln;
      #pragma unroll
      for (int r = 0; r < 8; ++r) op[(size_t)r * DD] = c[r];
    }
  }
}

extern "C" void kernel_launch(void* const* d_in, const int* in_sizes, int n_in,
                              void* d_out, int out_size, void* d_ws, size_t ws_size,
                              hipStream_t stream) {
  const float* x      = (const float*)d_in[0];
  const float* gate_w = (const float*)d_in[1];
  const float* gate_b = (const float*)d_in[2];
  const float* w1     = (const float*)d_in[3];
  const float* b1     = (const float*)d_in[4];
  const float* w2     = (const float*)d_in[5];
  float* out = (float*)d_out;

  // workspace layout (bf16): w1t 2MB | w2t 2MB | gwt 128KB
  char* ws = (char*)d_ws;
  __bf16* w1t = (__bf16*)(ws);
  __bf16* w2t = (__bf16*)(ws + (size_t)EE * HH * DD * 2);
  __bf16* gwt = (__bf16*)(ws + (size_t)EE * HH * DD * 4);

  const int nw = EE * HH * DD;  // 1,048,576
  moe_prep_w1<<<(nw + 255) / 256, 256, 0, stream>>>(w1, w1t);
  moe_prep_w2<<<(nw + 255) / 256, 256, 0, stream>>>(w2, w2t);
  moe_prep_gw<<<(EE * DD + 255) / 256, 256, 0, stream>>>(gate_w, gwt);

  moe_main<<<NTOK / MB, BLK, 0, stream>>>(x, gate_b, b1, gwt, w1t, w2t, out);
}
